// ConvolutionalAttention2D_43628277793389
// MI455X (gfx1250) — compile-verified, matrix operand order checked by reading, not run on hardware
//
#include <hip/hip_runtime.h>
#include <hip/hip_bf16.h>

// MI455X (gfx1250) fused linear-attention:
//   out = Wo * ( (phi(Wq x) phi(Wv x)^T) * phi(Wk x) ) + bo
// HBM-bound: ~768MB of traffic -> ~33us floor at 23.3TB/s. All matmuls except Wo*qv via
// v_wmma_f32_16x16x32_f16 so compute stays far under the memory roofline.

typedef __attribute__((ext_vector_type(16))) _Float16 v16h;
typedef __attribute__((ext_vector_type(8)))  _Float16 v8h;
typedef __attribute__((ext_vector_type(2)))  __fp16   v2fp;
typedef __attribute__((ext_vector_type(8)))  float    v8f;

#define B_    16
#define C_    64
#define HW_   65536
#define WGPB  128            // workgroups per batch
#define PXWG  (HW_ / WGPB)   // 512 pixels per workgroup
#define ITERS (PXWG / 64)    // 8 iterations of 64 pixels
#define LSTR  144            // LDS row stride (bytes): 16B aligned, 36-bank stride

__device__ __forceinline__ unsigned pkh(float a, float b) {
  v2fp t = __builtin_amdgcn_cvt_pkrtz(a, b);
  return __builtin_bit_cast(unsigned, t);
}

__device__ __forceinline__ float phi_elu1(float x) {  // elu(x)+1
  return x > 0.f ? x + 1.f : __expf(x);
}

// WMMA operand fetch: per-lane K-contiguous (16-bit A layout; B mirrors A of B^T).
// lane<16: row = lane, K {0..7} at +0 and {16..23} at +32
// lane>=16: row = lane-16, K {8..15} at +16 and {24..31} at +48
__device__ __forceinline__ v16h load_op(const unsigned char* base, int kc) {
  int lane = threadIdx.x & 31;
  const unsigned char* p = base + (lane & 15) * LSTR + kc * 64 + ((lane >> 4) << 4);
  v8h lo = *(const v8h*)p;
  v8h hi = *(const v8h*)(p + 32);
  v16h r;
#pragma unroll
  for (int i = 0; i < 8; ++i) { r[i] = lo[i]; r[i + 8] = hi[i]; }
  return r;
}

__device__ __forceinline__ v8f wmma16(v16h a, v16h b, v8f c) {
  return __builtin_amdgcn_wmma_f32_16x16x32_f16(false, a, false, b, (short)0, c,
                                                false, false);
}

// Stage one 64ch x 64px f32 tile -> f16, transposed [px][c] in LDS.
// Global reads are b128, 256B segments per channel row (coalesced).
__device__ __forceinline__ void stage_x(unsigned char* xt, const float* xb, int pxb) {
  int lane = threadIdx.x & 31, wave = threadIdx.x >> 5;
  int px4 = (lane & 15) * 4;
#pragma unroll
  for (int i = 0; i < 4; ++i) {
    int c = wave * 16 + (i * 2 + (lane >> 4)) * 2;   // even channel; lane packs c,c+1
    const float* r0 = xb + (size_t)c * HW_ + pxb + px4;
    float4 a = *(const float4*)r0;
    float4 b = *(const float4*)(r0 + HW_);
    __builtin_prefetch(r0 + 64, 0, 1);               // next 64-px chunk
#pragma unroll
    for (int j = 0; j < 4; ++j)
      *(unsigned*)(xt + (px4 + j) * LSTR + c * 2) = pkh((&a.x)[j], (&b.x)[j]);
  }
}

// 64x64 f32 weight -> f16 row-major in LDS (stride LSTR).
__device__ __forceinline__ void stage_w(unsigned char* dst, const float* W) {
  for (int e2 = threadIdx.x; e2 < 2048; e2 += 128) {
    int e = e2 * 2, o = e >> 6, c = e & 63;
    float2 w = *(const float2*)(W + e);
    *(unsigned*)(dst + o * LSTR + c * 2) = pkh(w.x, w.y);
  }
}

// ---------------- Pass 1: qv[b] = phi_q phi_v^T (atomic f32 accumulate) -------
__global__ __launch_bounds__(128) void k_qv(const float* __restrict__ x,
                                            const float* __restrict__ Wq,
                                            const float* __restrict__ Wv,
                                            float* __restrict__ qv) {
  __shared__ __align__(16) unsigned char lds[5 * 64 * LSTR];
  unsigned char* XT = lds;
  unsigned char* PQ = lds + 1 * 64 * LSTR;   // phi_q  [o][px]
  unsigned char* PV = lds + 2 * 64 * LSTR;   // phi_v  [o][px]
  unsigned char* WQ = lds + 3 * 64 * LSTR;
  unsigned char* WV = lds + 4 * 64 * LSTR;

  int b = blockIdx.x / WGPB, px0 = (blockIdx.x % WGPB) * PXWG;
  int lane = threadIdx.x & 31, wave = threadIdx.x >> 5;
  stage_w(WQ, Wq);
  stage_w(WV, Wv);
  const float* xb = x + (size_t)b * C_ * HW_;

  v8f acc[4];                                // this wave's 16cq x 64cv block
#pragma unroll
  for (int j = 0; j < 4; ++j) acc[j] = (v8f){};

  for (int it = 0; it < ITERS; ++it) {
    stage_x(XT, xb, px0 + it * 64);
    __syncthreads();
    // GEMM1: D[px][o] = x_tile[px][c] * W[o][c]   (wave owns 16 px)
    v16h ax0 = load_op(XT + wave * 16 * LSTR, 0);
    v16h ax1 = load_op(XT + wave * 16 * LSTR, 1);
#pragma unroll
    for (int ot = 0; ot < 4; ++ot) {
      v8f dq = (v8f){}, dv = (v8f){};
      dq = wmma16(ax0, load_op(WQ + ot * 16 * LSTR, 0), dq);
      dq = wmma16(ax1, load_op(WQ + ot * 16 * LSTR, 1), dq);
      dv = wmma16(ax0, load_op(WV + ot * 16 * LSTR, 0), dv);
      dv = wmma16(ax1, load_op(WV + ot * 16 * LSTR, 1), dv);
      int oo  = ot * 16 + (lane & 15);                 // N = o
      int pxo = wave * 16 + ((lane >> 4) << 3);        // M = px (+8 for hi lanes)
#pragma unroll
      for (int j = 0; j < 4; ++j) {
        *(unsigned*)(PQ + oo * LSTR + (pxo + 2 * j) * 2) =
            pkh(phi_elu1(dq[2 * j]), phi_elu1(dq[2 * j + 1]));
        *(unsigned*)(PV + oo * LSTR + (pxo + 2 * j) * 2) =
            pkh(phi_elu1(dv[2 * j]), phi_elu1(dv[2 * j + 1]));
      }
    }
    __syncthreads();
    // GEMM2: qv[cq][cv] += phi_q[cq][px] * phi_v[cv][px]   (K = 64 px)
    v16h aq0 = load_op(PQ + wave * 16 * LSTR, 0);
    v16h aq1 = load_op(PQ + wave * 16 * LSTR, 1);
#pragma unroll
    for (int j = 0; j < 4; ++j) {
      acc[j] = wmma16(aq0, load_op(PV + j * 16 * LSTR, 0), acc[j]);
      acc[j] = wmma16(aq1, load_op(PV + j * 16 * LSTR, 1), acc[j]);
    }
    __syncthreads();
  }

  float* qvb = qv + (size_t)b * 64 * 64;
#pragma unroll
  for (int j = 0; j < 4; ++j) {
    int cv = j * 16 + (lane & 15);
#pragma unroll
    for (int r = 0; r < 8; ++r) {
      int cq = wave * 16 + ((lane >> 4) << 3) + r;
      unsafeAtomicAdd(qvb + cq * 64 + cv, acc[j][r]);
    }
  }
}

// ---------------- Pass 2a: M[b] = (Wo * qv[b]) / 65536  in f16 ----------------
// qv ~ 7e4 overflows f16; exact power-of-2 pre-scale, undone after WMMA in f32.
__global__ __launch_bounds__(256) void k_m(const float* __restrict__ Wo,
                                           const float* __restrict__ qv,
                                           _Float16* __restrict__ mH) {
  int b = blockIdx.x;
  int o = threadIdx.x & 63, dg = threadIdx.x >> 6;
  const float* qvb = qv + (size_t)b * 4096;
#pragma unroll 4
  for (int i = 0; i < 16; ++i) {
    int d = dg * 16 + i;
    float s = 0.f;
    for (int c = 0; c < 64; ++c) s += Wo[o * 64 + c] * qvb[c * 64 + d];
    mH[(size_t)b * 4096 + o * 64 + d] = (_Float16)(s * (1.f / 65536.f));
  }
}

// ---------------- Pass 2b: out = M * phi(Wk x) * 65536 + bo -------------------
__global__ __launch_bounds__(128) void k_out(const float* __restrict__ x,
                                             const float* __restrict__ Wk,
                                             const _Float16* __restrict__ mH,
                                             const float* __restrict__ bo,
                                             float* __restrict__ out) {
  __shared__ __align__(16) unsigned char lds[4 * 64 * LSTR + 256];
  unsigned char* XT = lds;
  unsigned char* PK = lds + 1 * 64 * LSTR;   // phi_k [px][d]
  unsigned char* WK = lds + 2 * 64 * LSTR;
  unsigned char* MM = lds + 3 * 64 * LSTR;
  float* BO = (float*)(lds + 4 * 64 * LSTR);

  int b = blockIdx.x / WGPB, px0 = (blockIdx.x % WGPB) * PXWG;
  int lane = threadIdx.x & 31, wave = threadIdx.x >> 5;
  stage_w(WK, Wk);
  {
    const unsigned* src = (const unsigned*)(mH + (size_t)b * 4096);
    for (int e2 = threadIdx.x; e2 < 2048; e2 += 128) {
      int e = e2 * 2, o = e >> 6, d = e & 63;
      *(unsigned*)(MM + o * LSTR + d * 2) = src[e2];
    }
  }
  if (threadIdx.x < 64) BO[threadIdx.x] = bo[threadIdx.x];
  const float* xb = x + (size_t)b * C_ * HW_;

  for (int it = 0; it < ITERS; ++it) {
    int pxb = px0 + it * 64;
    stage_x(XT, xb, pxb);
    __syncthreads();
    // GEMM1: D[d][px] = Wk[d][c] * x_tile[px][c]; wave owns px tile -> no barrier
    v16h bx0 = load_op(XT + wave * 16 * LSTR, 0);
    v16h bx1 = load_op(XT + wave * 16 * LSTR, 1);
    int px = wave * 16 + (lane & 15);
#pragma unroll
    for (int dt = 0; dt < 4; ++dt) {
      v8f dk = (v8f){};
      dk = wmma16(load_op(WK + dt * 16 * LSTR, 0), bx0, dk);
      dk = wmma16(load_op(WK + dt * 16 * LSTR, 1), bx1, dk);
      int dbase = dt * 16 + ((lane >> 4) << 3);        // M = d (+8 hi lanes)
#pragma unroll
      for (int j = 0; j < 4; ++j)
        *(unsigned*)(PK + px * LSTR + (dbase + 2 * j) * 2) =
            pkh(phi_elu1(dk[2 * j]), phi_elu1(dk[2 * j + 1]));
    }
    // GEMM2: D[o][px] = M[o][d] * phi_k[px][d]; D layout stores coalesced to out
    v16h pk0 = load_op(PK + wave * 16 * LSTR, 0);
    v16h pk1 = load_op(PK + wave * 16 * LSTR, 1);
#pragma unroll
    for (int ot = 0; ot < 4; ++ot) {
      v8f oa = (v8f){};
      oa = wmma16(load_op(MM + ot * 16 * LSTR, 0), pk0, oa);
      oa = wmma16(load_op(MM + ot * 16 * LSTR, 1), pk1, oa);
      int obase = ot * 16 + ((lane >> 4) << 3);
      float* op = out + ((size_t)b * 64 + obase) * HW_ + pxb + px;
#pragma unroll
      for (int r = 0; r < 8; ++r)
        op[(size_t)r * HW_] = oa[r] * 65536.f + BO[obase + r];
    }
    __syncthreads();
  }
}

extern "C" void kernel_launch(void* const* d_in, const int* in_sizes, int n_in,
                              void* d_out, int out_size, void* d_ws, size_t ws_size,
                              hipStream_t stream) {
  (void)in_sizes; (void)n_in; (void)out_size; (void)ws_size;
  const float* x  = (const float*)d_in[0];
  const float* Wq = (const float*)d_in[1];
  const float* Wk = (const float*)d_in[2];
  const float* Wv = (const float*)d_in[3];
  const float* Wo = (const float*)d_in[4];
  const float* bo = (const float*)d_in[5];
  float* out = (float*)d_out;

  float*    qvF = (float*)d_ws;                                    // 16*64*64 f32
  _Float16* mH  = (_Float16*)((char*)d_ws + B_ * 64 * 64 * 4);     // 16*64*64 f16

  (void)hipMemsetAsync(d_ws, 0, B_ * 64 * 64 * sizeof(float), stream); // zero qv
  k_qv<<<B_ * WGPB, 128, 0, stream>>>(x, Wq, Wv, qvF);
  k_m<<<B_, 256, 0, stream>>>(Wo, qvF, mH);
  k_out<<<B_ * WGPB, 128, 0, stream>>>(x, Wk, mH, bo, out);
}
